// PairStr2Pair_3049426780674
// MI455X (gfx1250) — matrix operand order repaired and checked by reading, not compiled, not run on hardware
//
#include <hip/hip_runtime.h>
#include <hip/hip_bf16.h>
#include <math.h>

// ---------------------------------------------------------------------------
// PairStr2Pair for MI455X (gfx1250). GEMM stages on v_wmma_f32_16x16x32_bf16.
// LDS tiles laid out so each lane's WMMA fragment is two contiguous 16B runs
// (2x ds_load_b128). Tile fills in the GEMM/logits kernels use CDNA5 async
// global->LDS loads (global_load_async_to_lds_b128 + s_wait_asynccnt).
// ---------------------------------------------------------------------------

#define LQ   384        // L
#define DP   128        // d_pair
#define NH   4          // heads
#define DH   32         // d_hidden
#define DRBF 36
#define SQ   (LQ * LQ)  // 147456 rows
#define LDT  40         // LDS tile stride (elements): 80B rows keep 16B align

typedef __attribute__((ext_vector_type(16))) __bf16 bf16x16;
typedef __attribute__((ext_vector_type(8)))  float  f32x8;

__device__ __forceinline__ f32x8 wmma_bf16(bf16x16 a, bf16x16 b, f32x8 c) {
  return __builtin_amdgcn_wmma_f32_16x16x32_bf16(false, a, false, b, (short)0, c,
                                                 false, false);
}

// Async global->LDS copy of 16 bytes per lane (CDNA5 TDM-adjacent path).
// lds byte address = low 32 bits of the flat shared-pointer address.
__device__ __forceinline__ void async_copy_b128(const __bf16* gsrc, __bf16* ldst) {
  unsigned la = (unsigned)(size_t)(void*)ldst;
  unsigned long long ga = (unsigned long long)(size_t)gsrc;
  asm volatile("global_load_async_to_lds_b128 %0, %1, off"
               :: "v"(la), "v"(ga) : "memory");
}
__device__ __forceinline__ void async_wait0() {
  asm volatile("s_wait_asynccnt 0x0" ::: "memory");
}

// A fragment (16x32 bf16) from row-major tile; p = tile + (lane&15)*LDT.
// lanes 0-15: K {0..7,16..23}; lanes 16-31: K {8..15,24..31}  (2x b128)
__device__ __forceinline__ bf16x16 load_a_frag(const __bf16* p) {
  union { bf16x16 f; uint4 q[2]; } u;
  const int kb = (threadIdx.x & 16) >> 1;           // 0 or 8 elements
  u.q[0] = *(const uint4*)(p + kb);
  u.q[1] = *(const uint4*)(p + 16 + kb);
  return u.f;
}

// B fragment (32x16 bf16) from N-major tile; p = tile + (n_col+lane&15)*LDT.
// lanes 0-15: K {0..7,16..23}; lanes 16-31: K {8..15,24..31}  (2x b128)
__device__ __forceinline__ bf16x16 load_b_frag(const __bf16* p) {
  union { bf16x16 f; uint4 q[2]; } u;
  const int kb = (threadIdx.x & 16) >> 1;           // 0 or 8 elements
  u.q[0] = *(const uint4*)(p + kb);
  u.q[1] = *(const uint4*)(p + 16 + kb);
  return u.f;
}

// ---------------------------------------------------------------------------
// fp32 -> bf16 transposed weight convert: dst[n*K+k] = src[k*N+n]
__global__ void f2bfT_kernel(const float* __restrict__ src, __bf16* __restrict__ dst,
                             int K, int N) {
  int idx = blockIdx.x * blockDim.x + threadIdx.x;
  if (idx >= K * N) return;
  int n = idx / K, k = idx % K;
  dst[idx] = (__bf16)src[(size_t)k * N + n];
}

// ---------------------------------------------------------------------------
// rbf embed: out = relu(feat@W1 + b1) @ W2 + b2   (36->32->128), one wave/row
__global__ __launch_bounds__(128) void rbf_embed_kernel(
    const float* __restrict__ feat, const float* __restrict__ W1,
    const float* __restrict__ b1, const float* __restrict__ W2,
    const float* __restrict__ b2, float* __restrict__ out) {
  size_t row = (size_t)blockIdx.x * 4 + (threadIdx.x >> 5);
  int lane = threadIdx.x & 31;
  const float* f = feat + row * DRBF;
  float h = b1[lane];
#pragma unroll 4
  for (int r = 0; r < DRBF; ++r) h += f[r] * W1[r * DH + lane];
  h = fmaxf(h, 0.0f);
  float acc0 = b2[lane], acc1 = b2[lane + 32], acc2 = b2[lane + 64], acc3 = b2[lane + 96];
#pragma unroll
  for (int t = 0; t < 32; ++t) {
    float hv = __shfl(h, t, 32);
    const float* w = W2 + t * DP;
    acc0 += hv * w[lane];      acc1 += hv * w[lane + 32];
    acc2 += hv * w[lane + 64]; acc3 += hv * w[lane + 96];
  }
  float* o = out + row * DP;
  o[lane] = acc0; o[lane + 32] = acc1; o[lane + 64] = acc2; o[lane + 96] = acc3;
}

// ---------------------------------------------------------------------------
// LayerNorm(128) + cast to bf16, one wave per row
__global__ __launch_bounds__(256) void ln_cast_kernel(
    const float* __restrict__ x, const float* __restrict__ g,
    const float* __restrict__ b, __bf16* __restrict__ out) {
  size_t row = (size_t)blockIdx.x * 8 + (threadIdx.x >> 5);
  int lane = threadIdx.x & 31;
  const float* p = x + row * DP;
  float v[4];
  float s = 0.0f;
#pragma unroll
  for (int u = 0; u < 4; ++u) { v[u] = p[lane + 32 * u]; s += v[u]; }
#pragma unroll
  for (int sh = 16; sh; sh >>= 1) s += __shfl_xor(s, sh, 32);
  float mu = s * (1.0f / 128.0f);
  float var = 0.0f;
#pragma unroll
  for (int u = 0; u < 4; ++u) { float d = v[u] - mu; var += d * d; }
#pragma unroll
  for (int sh = 16; sh; sh >>= 1) var += __shfl_xor(var, sh, 32);
  float rstd = rsqrtf(var * (1.0f / 128.0f) + 1e-5f);
  __bf16* o = out + row * DP;
#pragma unroll
  for (int u = 0; u < 4; ++u) {
    int c = lane + 32 * u;
    o[c] = (__bf16)((v[u] - mu) * rstd * g[c] + b[c]);
  }
}

// ---------------------------------------------------------------------------
// Generic WMMA GEMM: C[M,N] = A[M,K](bf16) * W(bf16, PRE-TRANSPOSED [N][K]).
// Double-buffered LDS with async global->LDS fills; one barrier per K-step.
// mode 0: outb = bf16(scale*acc)          mode 1: outf = sigmoid(acc+bias)
// mode 2: outb = bf16(relu(acc+bias))     mode 3: outf = acc+bias+residual
__global__ __launch_bounds__(256) void gemm_bf16_kernel(
    const __bf16* __restrict__ A, const __bf16* __restrict__ Wt,
    const float* __restrict__ bias, const float* __restrict__ residual,
    float scale, int mode, int M, int K, int N,
    __bf16* __restrict__ outb, float* __restrict__ outf) {
  __shared__ __align__(16) __bf16 As[2][64 * LDT];
  __shared__ __align__(16) __bf16 Wn[2][64 * LDT];   // N-major
  const int m0 = blockIdx.x * 64, n0 = blockIdx.y * 64;
  const int t = threadIdx.x, lane = t & 31, wave = t >> 5;
  const int sr = (wave & 3) * 16, sc = (wave >> 2) * 32;
  const int lrow = lane & 15;
  const int fr = t >> 2, fkg = (t & 3) << 3;         // 64 rows x 32 k fills
  f32x8 acc0 = {}, acc1 = {};

  auto fill = [&](int buf, int kk) {
    async_copy_b128(A  + (size_t)(m0 + fr) * K + kk + fkg, As[buf] + fr * LDT + fkg);
    async_copy_b128(Wt + (size_t)(n0 + fr) * K + kk + fkg, Wn[buf] + fr * LDT + fkg);
  };
  fill(0, 0);
  async_wait0();
  __syncthreads();
  int cur = 0;
  for (int kk = 0; kk < K; kk += 32) {
    if (kk + 32 < K) fill(cur ^ 1, kk + 32);
    bf16x16 af = load_a_frag(As[cur] + (sr + lrow) * LDT);
    bf16x16 b0 = load_b_frag(Wn[cur] + (sc + lrow) * LDT);
    bf16x16 b1 = load_b_frag(Wn[cur] + (sc + 16 + lrow) * LDT);
    acc0 = wmma_bf16(af, b0, acc0);
    acc1 = wmma_bf16(af, b1, acc1);
    async_wait0();
    __syncthreads();
    cur ^= 1;
  }
  const int hi = lane >> 4, col = lane & 15;
#pragma unroll
  for (int st = 0; st < 2; ++st) {
    f32x8 acc = st ? acc1 : acc0;
#pragma unroll
    for (int v = 0; v < 8; ++v) {
      int gr = m0 + sr + v + hi * 8;
      int gc = n0 + sc + st * 16 + col;
      size_t o = (size_t)gr * N + gc;
      float val = acc[v] * scale;
      if (mode == 0)      outb[o] = (__bf16)val;
      else if (mode == 1) outf[o] = 1.0f / (1.0f + __expf(-(val + bias[gc])));
      else if (mode == 2) outb[o] = (__bf16)fmaxf(val + bias[gc], 0.0f);
      else                outf[o] = val + bias[gc] + residual[o];
    }
  }
}

// ---------------------------------------------------------------------------
// bb[h][i][j] = sum_c bn[storage(i,j),c] * Wb[c,h]
__global__ __launch_bounds__(256) void bias_proj_kernel(
    const __bf16* __restrict__ bn, const float* __restrict__ Wb,
    float* __restrict__ attn_bias, int T) {
  int h = blockIdx.y;
  size_t s = (size_t)blockIdx.x * blockDim.x + threadIdx.x;   // i*L+j
  int i = (int)(s / LQ), j = (int)(s % LQ);
  size_t srow = T ? ((size_t)j * LQ + i) : ((size_t)i * LQ + j);
  const __bf16* x = bn + srow * DP;
  float acc = 0.0f;
#pragma unroll 4
  for (int c = 0; c < DP; ++c) acc += (float)x[c] * Wb[c * NH + h];
  attn_bias[(size_t)h * SQ + s] = acc;
}

// ---------------------------------------------------------------------------
// attn[h][i][j] = sum_{n,k} q[n,i,h,k]*k[n,j,h,k] + bb[h][i][j]  (tied over n)
// q/k in storage order: storage(n,x) = T ? x*L+n : n*L+x
__global__ __launch_bounds__(256) void attn_logits_kernel(
    const __bf16* __restrict__ q, const __bf16* __restrict__ kmat,
    const float* __restrict__ attn_bias, float* __restrict__ attn, int T) {
  __shared__ __align__(16) __bf16 Aq[2][64 * LDT];   // [i][k] row-major
  __shared__ __align__(16) __bf16 Bk[2][64 * LDT];   // [j][k] == N-major for B
  const int i0 = blockIdx.x * 64, j0 = blockIdx.y * 64, h = blockIdx.z;
  const int t = threadIdx.x, lane = t & 31, wave = t >> 5;
  const int isub = (wave & 3) * 16, jsub = (wave >> 2) * 32;
  const int lrow = lane & 15;
  const int fr = t >> 2, fkg = (t & 3) << 3;
  f32x8 acc0 = {}, acc1 = {};

  auto fill = [&](int buf, int n) {
    int i = i0 + fr, j = j0 + fr;
    size_t srq = T ? ((size_t)i * LQ + n) : ((size_t)n * LQ + i);
    size_t srk = T ? ((size_t)j * LQ + n) : ((size_t)n * LQ + j);
    async_copy_b128(q    + srq * DP + h * DH + fkg, Aq[buf] + fr * LDT + fkg);
    async_copy_b128(kmat + srk * DP + h * DH + fkg, Bk[buf] + fr * LDT + fkg);
  };
  fill(0, 0);
  async_wait0();
  __syncthreads();
  int cur = 0;
  for (int n = 0; n < LQ; ++n) {
    if (n + 1 < LQ) {
      fill(cur ^ 1, n + 1);
      if (n + 8 < LQ) {   // global_prefetch_b8 a few tiles ahead
        int np = n + 8, i = i0 + fr;
        size_t srp = T ? ((size_t)i * LQ + np) : ((size_t)np * LQ + i);
        __builtin_prefetch(q + srp * DP + h * DH + fkg, 0, 1);
      }
    }
    bf16x16 af = load_a_frag(Aq[cur] + (isub + lrow) * LDT);
    bf16x16 b0 = load_b_frag(Bk[cur] + (jsub + lrow) * LDT);
    bf16x16 b1 = load_b_frag(Bk[cur] + (jsub + 16 + lrow) * LDT);
    acc0 = wmma_bf16(af, b0, acc0);
    acc1 = wmma_bf16(af, b1, acc1);
    async_wait0();
    __syncthreads();
    cur ^= 1;
  }
  const int hi = lane >> 4, col = lane & 15;
#pragma unroll
  for (int st = 0; st < 2; ++st) {
    f32x8 acc = st ? acc1 : acc0;
#pragma unroll
    for (int v = 0; v < 8; ++v) {
      int gi = i0 + isub + v + hi * 8;
      int gj = j0 + jsub + st * 16 + col;
      size_t o = (size_t)h * SQ + (size_t)gi * LQ + gj;
      attn[o] = acc[v] + attn_bias[o];
    }
  }
}

// ---------------------------------------------------------------------------
// softmax over j for each (h,i): one wave per row of 384 (contiguous)
__global__ __launch_bounds__(256) void softmax_kernel(float* __restrict__ attn) {
  int wid = blockIdx.x * 8 + (threadIdx.x >> 5);
  int lane = threadIdx.x & 31;
  int h = wid / LQ, i = wid % LQ;
  float* row = attn + (size_t)h * SQ + (size_t)i * LQ;
  float m = -1e30f;
  for (int j = lane; j < LQ; j += 32) m = fmaxf(m, row[j]);
#pragma unroll
  for (int s = 16; s; s >>= 1) m = fmaxf(m, __shfl_xor(m, s, 32));
  float sum = 0.0f;
  for (int j = lane; j < LQ; j += 32) sum += __expf(row[j] - m);
#pragma unroll
  for (int s = 16; s; s >>= 1) sum += __shfl_xor(sum, s, 32);
  float inv = 1.0f / sum;
  for (int j = lane; j < LQ; j += 32) row[j] = __expf(row[j] - m) * inv;
}

// ---------------------------------------------------------------------------
// out[n,i,h,d] = sum_j attn[h][i][j] * v[n,j,h,d]  (per (n,h): 384x384 x 384x32)
__global__ __launch_bounds__(256) void attn_v_kernel(
    const float* __restrict__ attn, const __bf16* __restrict__ vmat,
    float* __restrict__ outp, int T) {
  __shared__ __align__(16) __bf16 At[2][64 * LDT];   // [i][j] row-major
  __shared__ __align__(16) __bf16 Vt[2][32 * LDT];   // [d][j]: N-major for B
  const int i0 = blockIdx.x * 64, h = blockIdx.y, n = blockIdx.z;
  const int t = threadIdx.x, lane = t & 31, wave = t >> 5;
  const int isub = (wave & 3) * 16, dsub = (wave >> 2) * 16;
  const int lrow = lane & 15;
  f32x8 acc = {};

  auto fill = [&](int buf, int jj) {
    {   // attn tile 64(i) x 32(j): vector f32 loads, convert, b128 LDS store
      int r = t >> 2, jg = (t & 3) << 3;
      const float* src = attn + (size_t)h * SQ + (size_t)(i0 + r) * LQ + jj + jg;
      float4 a = *(const float4*)src;
      float4 b = *(const float4*)(src + 4);
      union { uint4 q; __bf16 e[8]; } u;
      u.e[0] = (__bf16)a.x; u.e[1] = (__bf16)a.y; u.e[2] = (__bf16)a.z; u.e[3] = (__bf16)a.w;
      u.e[4] = (__bf16)b.x; u.e[5] = (__bf16)b.y; u.e[6] = (__bf16)b.z; u.e[7] = (__bf16)b.w;
      *(uint4*)(At[buf] + r * LDT + jg) = u.q;
    }
    {   // v tile -> [d][j] (transpose-on-store, 4 scalar b16 stores)
      int jr = t >> 3, dg = (t & 7) << 2;
      size_t srow = T ? ((size_t)(jj + jr) * LQ + n) : ((size_t)n * LQ + jj + jr);
      union { uint2 q; __bf16 e[4]; } u;
      u.q = *(const uint2*)(vmat + srow * DP + h * DH + dg);
#pragma unroll
      for (int x = 0; x < 4; ++x) Vt[buf][(dg + x) * LDT + jr] = u.e[x];
    }
  };
  fill(0, 0);
  __syncthreads();
  int cur = 0;
  for (int jj = 0; jj < LQ; jj += 32) {
    if (jj + 32 < LQ) fill(cur ^ 1, jj + 32);
    bf16x16 af  = load_a_frag(At[cur] + (isub + lrow) * LDT);
    bf16x16 bfr = load_b_frag(Vt[cur] + (dsub + lrow) * LDT);
    acc = wmma_bf16(af, bfr, acc);
    __syncthreads();
    cur ^= 1;
  }
  const int hi = lane >> 4, col = lane & 15;
#pragma unroll
  for (int vv = 0; vv < 8; ++vv) {
    int gi = i0 + isub + vv + hi * 8;
    size_t srow = T ? ((size_t)gi * LQ + n) : ((size_t)n * LQ + gi);
    outp[srow * DP + h * DH + dsub + col] = acc[vv];
  }
}

// ---------------------------------------------------------------------------
// gated = bf16(g * out_pre)
__global__ void gate_mul_kernel(const float* __restrict__ g, const float* __restrict__ x,
                                __bf16* __restrict__ out, size_t n) {
  size_t i = (size_t)blockIdx.x * blockDim.x + threadIdx.x;
  if (i < n) out[i] = (__bf16)(g[i] * x[i]);
}

// ===========================================================================
extern "C" void kernel_launch(void* const* d_in, const int* in_sizes, int n_in,
                              void* d_out, int out_size, void* d_ws, size_t ws_size,
                              hipStream_t stream) {
  const float* pair_in  = (const float*)d_in[0];
  const float* rbf_feat = (const float*)d_in[1];
  const float* emb_W = (const float*)d_in[2];
  const float* emb_b = (const float*)d_in[3];
  const float* prj_W = (const float*)d_in[4];
  const float* prj_b = (const float*)d_in[5];
  // row params d_in[6..17], col params d_in[18..29], ff params d_in[30..35]

  const size_t R = (size_t)SQ * DP;
  char* w = (char*)d_ws;
  auto alloc = [&](size_t bytes) -> void* {
    void* p = (void*)w;
    w += (bytes + 255) & ~(size_t)255;
    return p;
  };
  float*  rbf    = (float*)alloc(R * 4);
  __bf16* pn     = (__bf16*)alloc(R * 2);
  __bf16* bn     = (__bf16*)alloc(R * 2);
  __bf16* qb     = (__bf16*)alloc(R * 2);
  __bf16* kb     = (__bf16*)alloc(R * 2);
  __bf16* vb     = (__bf16*)alloc(R * 2);
  float*  gf     = (float*)alloc(R * 4);
  float*  tmp    = (float*)alloc(R * 4);
  __bf16* gated  = (__bf16*)alloc(R * 2);
  float*  attn   = (float*)alloc((size_t)SQ * NH * 4);
  float*  attn_b = (float*)alloc((size_t)SQ * NH * 4);
  __bf16* hidden = (__bf16*)alloc((size_t)SQ * 256 * 2);
  __bf16* wbf[2][5];   // per-pass Wq,Wk,Wv,Wg,Wo (transposed [N][K])
  for (int p = 0; p < 2; ++p)
    for (int m = 0; m < 5; ++m) wbf[p][m] = (__bf16*)alloc(DP * DP * 2);
  __bf16* w1bf = (__bf16*)alloc((size_t)DP * 256 * 2);
  __bf16* w2bf = (__bf16*)alloc((size_t)256 * DP * 2);

  float* out = (float*)d_out;

  auto cvtT = [&](const float* src, __bf16* dst, int K, int N) {
    f2bfT_kernel<<<(K * N + 255) / 256, 256, 0, stream>>>(src, dst, K, N);
  };
  for (int p = 0; p < 2; ++p) {
    int base = 6 + p * 12;
    cvtT((const float*)d_in[base + 4],  wbf[p][0], DP, DP);   // Wq
    cvtT((const float*)d_in[base + 5],  wbf[p][1], DP, DP);   // Wk
    cvtT((const float*)d_in[base + 6],  wbf[p][2], DP, DP);   // Wv
    cvtT((const float*)d_in[base + 8],  wbf[p][3], DP, DP);   // Wg
    cvtT((const float*)d_in[base + 10], wbf[p][4], DP, DP);   // Wo
  }
  cvtT((const float*)d_in[32], w1bf, DP, 256);
  cvtT((const float*)d_in[34], w2bf, 256, DP);

  rbf_embed_kernel<<<SQ / 4, 128, 0, stream>>>(rbf_feat, emb_W, emb_b, prj_W, prj_b, rbf);

  const float qscale = 0.17677669529663687f;   // 1/sqrt(32)
  const float kscale = 1.0f / (float)LQ;
  const dim3 gproj(SQ / 64, DP / 64);

  // --- two axial attention passes: p=0 row (T=1), p=1 col (T=0) ---
  for (int p = 0; p < 2; ++p) {
    const int base = 6 + p * 12;
    const int T = (p == 0) ? 1 : 0;
    const float* ln_pg = (const float*)d_in[base + 0];
    const float* ln_pb = (const float*)d_in[base + 1];
    const float* ln_bg = (const float*)d_in[base + 2];
    const float* ln_bb = (const float*)d_in[base + 3];
    const float* Wb_f  = (const float*)d_in[base + 7];
    const float* bg    = (const float*)d_in[base + 9];
    const float* bo    = (const float*)d_in[base + 11];
    const float* pair_cur = (p == 0) ? pair_in : out;

    ln_cast_kernel<<<SQ / 8, 256, 0, stream>>>(pair_cur, ln_pg, ln_pb, pn);
    ln_cast_kernel<<<SQ / 8, 256, 0, stream>>>(rbf, ln_bg, ln_bb, bn);

    gemm_bf16_kernel<<<gproj, 256, 0, stream>>>(pn, wbf[p][0], nullptr, nullptr,
        qscale, 0, SQ, DP, DP, qb, nullptr);
    gemm_bf16_kernel<<<gproj, 256, 0, stream>>>(pn, wbf[p][1], nullptr, nullptr,
        kscale, 0, SQ, DP, DP, kb, nullptr);
    gemm_bf16_kernel<<<gproj, 256, 0, stream>>>(pn, wbf[p][2], nullptr, nullptr,
        1.0f, 0, SQ, DP, DP, vb, nullptr);
    gemm_bf16_kernel<<<gproj, 256, 0, stream>>>(pn, wbf[p][3], bg, nullptr,
        1.0f, 1, SQ, DP, DP, nullptr, gf);

    bias_proj_kernel<<<dim3(SQ / 256, NH), 256, 0, stream>>>(bn, Wb_f, attn_b, T);

    attn_logits_kernel<<<dim3(LQ / 64, LQ / 64, NH), 256, 0, stream>>>(qb, kb, attn_b, attn, T);
    softmax_kernel<<<LQ * NH / 8, 256, 0, stream>>>(attn);
    attn_v_kernel<<<dim3(LQ / 64, NH, LQ), 256, 0, stream>>>(attn, vb, tmp, T);

    gate_mul_kernel<<<(R + 255) / 256, 256, 0, stream>>>(gf, tmp, gated, R);
    gemm_bf16_kernel<<<gproj, 256, 0, stream>>>(gated, wbf[p][4], bo, pair_cur,
        1.0f, 3, SQ, DP, DP, nullptr, out);   // out = pair + attn_out
  }

  // --- feed-forward: out += W2(relu(W1(LN(out)))) ---
  const float* ff_g  = (const float*)d_in[30];
  const float* ff_b  = (const float*)d_in[31];
  const float* ff_b1 = (const float*)d_in[33];
  const float* ff_b2 = (const float*)d_in[35];
  ln_cast_kernel<<<SQ / 8, 256, 0, stream>>>(out, ff_g, ff_b, pn);
  gemm_bf16_kernel<<<dim3(SQ / 64, 256 / 64), 256, 0, stream>>>(pn, w1bf, ff_b1, nullptr,
      1.0f, 2, SQ, DP, 256, hidden, nullptr);
  gemm_bf16_kernel<<<dim3(SQ / 64, DP / 64), 256, 0, stream>>>(hidden, w2bf, ff_b2, out,
      1.0f, 3, SQ, 256, DP, nullptr, out);
}
